// MoE_61091614819116
// MI455X (gfx1250) — matrix operand order repaired and checked by reading, not compiled, not run on hardware
//
#include <hip/hip_runtime.h>

#define SDIM 8192
#define DDIM 1024
#define EEXP 8
#define TOPK 2

typedef __bf16 bf16_t;
typedef bf16_t v8bf  __attribute__((ext_vector_type(8)));
typedef bf16_t v16bf __attribute__((ext_vector_type(16)));
typedef float  v8f   __attribute__((ext_vector_type(8)));

__device__ __forceinline__ unsigned short f2bf(float f) {
    unsigned int u = __float_as_uint(f);
    u += 0x7fffu + ((u >> 16) & 1u);   // round-to-nearest-even
    return (unsigned short)(u >> 16);
}

// ---------------- init: zero output + per-expert counts ----------------
__global__ void moe_init_kernel(float* __restrict__ out, int* __restrict__ counts) {
    long long i = (long long)blockIdx.x * blockDim.x + threadIdx.x;
    if (i < (long long)SDIM * DDIM) out[i] = 0.0f;
    if (i < EEXP) counts[i] = 0;
}

// ---------------- f32 -> bf16 conversion (vectorized x4) ----------------
__global__ void moe_cvt_kernel(const float4* __restrict__ src,
                               ushort4* __restrict__ dst, int n4) {
    int i = blockIdx.x * blockDim.x + threadIdx.x;
    if (i >= n4) return;
    float4 v = src[i];
    ushort4 o;
    o.x = f2bf(v.x); o.y = f2bf(v.y); o.z = f2bf(v.z); o.w = f2bf(v.w);
    dst[i] = o;
}

// ---------------- gating: one wave32 per token ----------------
__global__ void moe_gate_kernel(const float* __restrict__ x,
                                const float* __restrict__ gw,
                                const float* __restrict__ gb,
                                int* __restrict__ topk_idx,
                                float* __restrict__ topk_w,
                                int* __restrict__ counts) {
    const int wv   = threadIdx.x >> 5;
    const int lane = threadIdx.x & 31;
    const int s    = blockIdx.x * 8 + wv;
    if (s >= SDIM) return;

    const float4* xr = (const float4*)(x + (long long)s * DDIM);
    float acc[EEXP];
#pragma unroll
    for (int e = 0; e < EEXP; ++e) acc[e] = 0.0f;

    // D/4 = 256 float4 per row; 32 lanes -> 8 strided iterations
    for (int it = 0; it < 8; ++it) {
        int idx = it * 32 + lane;
        float4 xv = xr[idx];
#pragma unroll
        for (int e = 0; e < EEXP; ++e) {
            float4 w4 = ((const float4*)(gw + (long long)e * DDIM))[idx];
            acc[e] += xv.x * w4.x + xv.y * w4.y + xv.z * w4.z + xv.w * w4.w;
        }
    }
    // butterfly reduce across the wave32
#pragma unroll
    for (int e = 0; e < EEXP; ++e) {
#pragma unroll
        for (int off = 16; off >= 1; off >>= 1)
            acc[e] += __shfl_xor(acc[e], off, 32);
        acc[e] += gb[e];
    }
    // softmax over 8 experts (redundant per lane, cheap)
    float mx = acc[0];
#pragma unroll
    for (int e = 1; e < EEXP; ++e) mx = fmaxf(mx, acc[e]);
    float p[EEXP], denom = 0.0f;
#pragma unroll
    for (int e = 0; e < EEXP; ++e) { p[e] = __expf(acc[e] - mx); denom += p[e]; }
    // top-2 (first-index-wins on ties, matching jax top_k)
    int i0 = 0;
#pragma unroll
    for (int e = 1; e < EEXP; ++e) if (p[e] > p[i0]) i0 = e;
    int i1 = (i0 == 0) ? 1 : 0;
#pragma unroll
    for (int e = 0; e < EEXP; ++e) if (e != i0 && p[e] > p[i1]) i1 = e;

    if (lane == 0) {
        topk_idx[s * 2 + 0] = i0;
        topk_idx[s * 2 + 1] = i1;
        topk_w[s * 2 + 0] = p[i0] / denom;
        topk_w[s * 2 + 1] = p[i1] / denom;
        atomicAdd(&counts[i0], 1);
        atomicAdd(&counts[i1], 1);
    }
}

// ---------------- prefix sum over E=8 + cursor reset ----------------
__global__ void moe_prefix_kernel(const int* __restrict__ counts,
                                  int* __restrict__ offsets,
                                  int* __restrict__ cursors) {
    if (threadIdx.x == 0) {
        int acc = 0;
        for (int e = 0; e < EEXP; ++e) { offsets[e] = acc; acc += counts[e]; }
    }
    if (threadIdx.x < EEXP) cursors[threadIdx.x] = 0;
}

// ---------------- scatter (token,slot) into per-expert lists ----------------
__global__ void moe_scatter_kernel(const int* __restrict__ topk_idx,
                                   const float* __restrict__ topk_w,
                                   const int* __restrict__ offsets,
                                   int* __restrict__ cursors,
                                   int* __restrict__ token_ids,
                                   float* __restrict__ slot_w) {
    int t = blockIdx.x * blockDim.x + threadIdx.x;
    if (t >= SDIM * TOPK) return;
    int e = topk_idx[t];
    int pos = atomicAdd(&cursors[e], 1);
    token_ids[offsets[e] + pos] = t >> 1;
    slot_w[offsets[e] + pos]   = topk_w[t];
}

// ---------------- grouped bf16 WMMA GEMM: Y = X_g @ W_e^T ----------------
// Block: 256 thr = 8 waves (2 M x 4 N); wave tile 32x64; block tile 64x256.
// K loop fully unrolled: each row's K sweep spans only 2 KB, so every
// global_load_b128 uses one fixed base address + an immediate IOFFSET.
// No per-iteration address math, no buffer-rotation moves, no WAR nops.
__global__ __launch_bounds__(256)
void moe_gemm_kernel(const unsigned short* __restrict__ xb,   // bf16 bits [S][D]
                     const unsigned short* __restrict__ wb,   // bf16 bits [E][D][D]
                     const float* __restrict__ expert_b,      // [E][D]
                     const int* __restrict__ counts,
                     const int* __restrict__ offsets,
                     const int* __restrict__ token_ids,
                     const float* __restrict__ slot_w,
                     float* __restrict__ out) {
    const int e  = blockIdx.z;
    const int ne = counts[e];
    const int mBase = blockIdx.y * 64;
    if (mBase >= ne) return;
    const int nBase = blockIdx.x * 256;

    __shared__ int   s_id[64];
    __shared__ float s_w[64];
    const int t = threadIdx.x;
    if (t < 64) {
        int slot = mBase + t;
        if (slot < ne) {
            s_id[t] = token_ids[offsets[e] + slot];
            s_w[t]  = slot_w[offsets[e] + slot];
        } else {
            s_id[t] = 0;
            s_w[t]  = 0.0f;    // zero weight => zero contribution, no guard needed
        }
    }
    __syncthreads();

    const int lane = t & 31;
    const int wv   = t >> 5;
    const int wm   = wv >> 2;      // 0..1  (M wave)
    const int wn   = wv & 3;       // 0..3  (N wave)
    const int half = lane >> 4;    // 0/1
    const int l16  = lane & 15;

    // Lane base pointers. A (k = 16*half + s, same order as B): lane row m=l16, one 32B
    // chunk per K step at +16*half. B: B[k][n]=W[n][k], lane
    // col n=l16, one 32B chunk per K step at +16*half.
    const bf16_t* aPtr[2];
#pragma unroll
    for (int tm = 0; tm < 2; ++tm) {
        int mLoc = wm * 32 + tm * 16 + l16;
        aPtr[tm] = (const bf16_t*)xb + (long long)s_id[mLoc] * DDIM + 16 * half;
    }
    const bf16_t* bPtr[4];
#pragma unroll
    for (int tn = 0; tn < 4; ++tn) {
        int n = nBase + wn * 64 + tn * 16 + l16;
        bPtr[tn] = (const bf16_t*)wb + ((long long)e * DDIM + n) * DDIM + 16 * half;
    }

    // Per-lane epilogue metadata in registers (branchless epilogue).
    int   rid[2][8];
    float rw [2][8];
#pragma unroll
    for (int tm = 0; tm < 2; ++tm)
#pragma unroll
        for (int r = 0; r < 8; ++r) {
            int mLoc = wm * 32 + tm * 16 + half * 8 + r;
            rid[tm][r] = s_id[mLoc];
            rw [tm][r] = s_w[mLoc];
        }

    v8f c[2][4];
#pragma unroll
    for (int tm = 0; tm < 2; ++tm)
#pragma unroll
        for (int tn = 0; tn < 4; ++tn)
#pragma unroll
            for (int r = 0; r < 8; ++r) c[tm][tn][r] = 0.0f;

    // ---- fully unrolled K loop: all loads use immediate offsets ----
#pragma unroll
    for (int k = 0; k < DDIM; k += 32) {
        v16bf a[2], b[4];
#pragma unroll
        for (int tm = 0; tm < 2; ++tm) {
            a[tm] = *(const v16bf*)(aPtr[tm] + k);
        }
#pragma unroll
        for (int tn = 0; tn < 4; ++tn)
            b[tn] = *(const v16bf*)(bPtr[tn] + k);

#pragma unroll
        for (int tm = 0; tm < 2; ++tm)
#pragma unroll
            for (int tn = 0; tn < 4; ++tn)
                c[tm][tn] = __builtin_amdgcn_wmma_f32_16x16x32_bf16(
                    false, a[tm], false, b[tn], (short)0, c[tm][tn], false, false);
    }

    // ---- branchless epilogue: out[s] += gate_w * (dot + bias) ----
    // Exactly two atomic adds per output element grid-wide (fp add is
    // commutative -> bitwise deterministic result).
#pragma unroll
    for (int tn = 0; tn < 4; ++tn) {
        int n = nBase + wn * 64 + tn * 16 + l16;
        float bias = expert_b[e * DDIM + n];
#pragma unroll
        for (int tm = 0; tm < 2; ++tm) {
#pragma unroll
            for (int r = 0; r < 8; ++r) {
                float v = (c[tm][tn][r] + bias) * rw[tm][r];
                unsafeAtomicAdd(out + ((long long)rid[tm][r] * DDIM + n), v);
            }
        }
    }
}

// ---------------- host launcher ----------------
extern "C" void kernel_launch(void* const* d_in, const int* in_sizes, int n_in,
                              void* d_out, int out_size, void* d_ws, size_t ws_size,
                              hipStream_t stream) {
    const float* x  = (const float*)d_in[0];
    const float* gw = (const float*)d_in[1];
    const float* gb = (const float*)d_in[2];
    const float* ew = (const float*)d_in[3];
    const float* eb = (const float*)d_in[4];
    float* out = (float*)d_out;

    char* ws = (char*)d_ws;
    size_t off = 0;
    auto alloc = [&](size_t bytes) -> void* {
        void* p = ws + off;
        off = (off + bytes + 255) & ~(size_t)255;
        return p;
    };
    unsigned short* xb  = (unsigned short*)alloc((size_t)SDIM * DDIM * 2);
    unsigned short* wbf = (unsigned short*)alloc((size_t)EEXP * DDIM * DDIM * 2);
    int*   counts    = (int*)  alloc(EEXP * sizeof(int));
    int*   offsets   = (int*)  alloc(EEXP * sizeof(int));
    int*   cursors   = (int*)  alloc(EEXP * sizeof(int));
    int*   topk_idx  = (int*)  alloc((size_t)SDIM * TOPK * sizeof(int));
    float* topk_w    = (float*)alloc((size_t)SDIM * TOPK * sizeof(float));
    int*   token_ids = (int*)  alloc((size_t)SDIM * TOPK * sizeof(int));
    float* slot_w    = (float*)alloc((size_t)SDIM * TOPK * sizeof(float));

    // 1) zero output + counts
    {
        long long n = (long long)SDIM * DDIM;
        int blocks = (int)((n + 255) / 256);
        moe_init_kernel<<<blocks, 256, 0, stream>>>(out, counts);
    }
    // 2) convert x and expert_w to bf16
    {
        int n4 = SDIM * DDIM / 4;
        moe_cvt_kernel<<<(n4 + 255) / 256, 256, 0, stream>>>(
            (const float4*)x, (ushort4*)xb, n4);
        int w4 = EEXP * DDIM * DDIM / 4;
        moe_cvt_kernel<<<(w4 + 255) / 256, 256, 0, stream>>>(
            (const float4*)ew, (ushort4*)wbf, w4);
    }
    // 3) gating (one wave per token, 8 waves per block)
    moe_gate_kernel<<<SDIM / 8, 256, 0, stream>>>(x, gw, gb, topk_idx, topk_w, counts);
    // 4) prefix sum + cursor reset
    moe_prefix_kernel<<<1, 32, 0, stream>>>(counts, offsets, cursors);
    // 5) scatter tokens into per-expert lists
    moe_scatter_kernel<<<(SDIM * TOPK + 255) / 256, 256, 0, stream>>>(
        topk_idx, topk_w, offsets, cursors, token_ids, slot_w);
    // 6) grouped WMMA GEMM; grid covers worst case (one expert gets all tokens)
    {
        dim3 grid(DDIM / 256, SDIM / 64, EEXP);   // (4, 128, 8); extras exit early
        moe_gemm_kernel<<<grid, 256, 0, stream>>>(
            xb, wbf, eb, counts, offsets, token_ids, slot_w, out);
    }
}
